// Recurrence_82085414961833
// MI455X (gfx1250) — matrix operand order repaired and checked by reading, not compiled, not run on hardware
//
#include <hip/hip_runtime.h>
#include <hip/hip_bf16.h>
#include <math.h>

// Problem constants (match reference)
#define TT      64
#define NBATCH  256
#define HH      1024
#define CDIM    256
#define LLN     16
#define NAC     33
#define XDIM    (CDIM + 2*HH)        // 2304
#define ODIM    (4 + HH + 2*NAC)     // 1094

typedef __bf16 bf16;
typedef __attribute__((ext_vector_type(16))) __bf16 v16bf;
typedef __attribute__((ext_vector_type(8)))  __bf16 v8bf;
typedef __attribute__((ext_vector_type(8)))  float  v8f;

// ---------------------------------------------------------------------------
// WMMA helpers (wave32, CDNA5 v_wmma_f32_16x16x32_bf16)
// ---------------------------------------------------------------------------
__device__ __forceinline__ v8f wmma_bf16(v16bf a, v16bf b, v8f c) {
    // 8 args: (neg_a, A, neg_b, B, c_mod, C, reuse_a, reuse_b)
    return __builtin_amdgcn_wmma_f32_16x16x32_bf16(
        false, a, false, b, (short)0, c, false, false);
}

// A-fragment (16x32 bf16 tile, row-major source). rowp already points at this
// lane's row (row = r0 + (lane&15)). Lanes 0-15 hold
// K = k0+0..15, lanes 16-31 hold K = k0+16..31 (matches B fragment).
__device__ __forceinline__ v16bf load_a_frag(const bf16* rowp, int k0) {
    const int lane = threadIdx.x & 31;
    const bf16* p = rowp + k0 + ((lane >> 4) << 4);
    v8bf lo = *(const v8bf*)(p);
    v8bf hi = *(const v8bf*)(p + 8);
    return __builtin_shufflevector(lo, hi,
        0,1,2,3,4,5,6,7,8,9,10,11,12,13,14,15);
}

// B-fragment for  X @ W^T : B[k][j] = W[j][k], W row-major [outdim, ldw].
// Contiguous k order, same as A: lane holds column j = col0 + (lane&15); lanes 0-15 carry
// K = k0+0..15, lanes 16-31 carry K = k0+16..31 (contiguous along W's row).
__device__ __forceinline__ v16bf load_b_fragT(const bf16* W, int ldw,
                                              int col0, int k0) {
    const int lane = threadIdx.x & 31;
    const bf16* p = W + (size_t)(col0 + (lane & 15)) * ldw
                      + k0 + ((lane >> 4) << 4);
    return *(const v16bf*)p;
}

__device__ __forceinline__ float sigmoidf(float x) {
    return 1.0f / (1.0f + __expf(-x));
}

// C/D f32 16x16 layout: acc[i] of lane l is element
// (row = i + 8*(l>>4), col = l&15)
__device__ __forceinline__ int c_row(int i) {
    return i + (((threadIdx.x & 31) >> 4) << 3);
}

// ---------------------------------------------------------------------------
// Fused GRU-gate tile: 6 accumulators over K=HH, 6 WMMAs per K-step.
// ---------------------------------------------------------------------------
struct GruAcc { v8f xr, xz, xn, hr, hz, hn; };

__device__ __forceinline__ GruAcc gru_gates_tile(const bf16* xrow,
                                                 const bf16* hrow,
                                                 const bf16* Wih,
                                                 const bf16* Whh,
                                                 int j0) {
    GruAcc g;
    g.xr = {}; g.xz = {}; g.xn = {}; g.hr = {}; g.hz = {}; g.hn = {};
    for (int k0 = 0; k0 < HH; k0 += 32) {
        v16bf ax = load_a_frag(xrow, k0);
        v16bf ah = load_a_frag(hrow, k0);
        if (k0 + 32 < HH) {
            __builtin_prefetch(Wih + (size_t)j0 * HH + k0 + 32, 0, 1);
        }
        g.xr = wmma_bf16(ax, load_b_fragT(Wih, HH, j0,        k0), g.xr);
        g.xz = wmma_bf16(ax, load_b_fragT(Wih, HH, j0 + HH,   k0), g.xz);
        g.xn = wmma_bf16(ax, load_b_fragT(Wih, HH, j0 + 2*HH, k0), g.xn);
        g.hr = wmma_bf16(ah, load_b_fragT(Whh, HH, j0,        k0), g.hr);
        g.hz = wmma_bf16(ah, load_b_fragT(Whh, HH, j0 + HH,   k0), g.hz);
        g.hn = wmma_bf16(ah, load_b_fragT(Whh, HH, j0 + 2*HH, k0), g.hn);
    }
    return g;
}

// ---------------------------------------------------------------------------
// Prep kernels
// ---------------------------------------------------------------------------
__global__ void k_f32_to_bf16(bf16* __restrict__ dst,
                              const float* __restrict__ src, int n) {
    int i = blockIdx.x * blockDim.x + threadIdx.x;
    if (i < n) dst[i] = (bf16)src[i];
}

// Pack a/p/v heads into one zero-padded [128 x 1024] bf16 matrix + bias[128].
// rows 0..32 = aW, rows 48..80 = pW, row 96 = vW, rest zero.
__global__ void k_pack_heads(bf16* __restrict__ Wh, float* __restrict__ bh,
                             const float* __restrict__ aW,
                             const float* __restrict__ ab,
                             const float* __restrict__ pW,
                             const float* __restrict__ pb,
                             const float* __restrict__ vW,
                             const float* __restrict__ vb) {
    int i = blockIdx.x * blockDim.x + threadIdx.x;
    if (i < 128 * HH) {
        int r = i >> 10, c = i & (HH - 1);
        float v = 0.0f;
        if (r < NAC)                     v = aW[r * HH + c];
        else if (r >= 48 && r < 48+NAC)  v = pW[(r - 48) * HH + c];
        else if (r == 96)                v = vW[c];
        Wh[i] = (bf16)v;
    }
    if (i < 128) {
        float v = 0.0f;
        if (i < NAC)                     v = ab[i];
        else if (i >= 48 && i < 48+NAC)  v = pb[i - 48];
        else if (i == 96)                v = vb[0];
        bh[i] = v;
    }
}

// ---------------------------------------------------------------------------
// Phase 1: BiGRU over all 16 cyclic rolls. One block per (roll, dir) chain,
// 8 waves. Hidden state ping-pongs in workspace (bf16); embedding gather is
// fused into the A-fragment row pointers.
// ---------------------------------------------------------------------------
__global__ __launch_bounds__(256, 1)
void bigru_kernel(const bf16* __restrict__ embed_bf,
                  const int*  __restrict__ lines,
                  const bf16* __restrict__ Wih_f, const bf16* __restrict__ Whh_f,
                  const bf16* __restrict__ Wih_b, const bf16* __restrict__ Whh_b,
                  const float* __restrict__ bih_f, const float* __restrict__ bhh_f,
                  const float* __restrict__ bih_b, const float* __restrict__ bhh_b,
                  bf16* __restrict__ hbuf,  // [32 chains][2][256*1024]
                  bf16* __restrict__ G)     // [16][256][2048]
{
    const int chain = blockIdx.x;
    const int roll = chain >> 1, dir = chain & 1;
    const bf16*  Wih = dir ? Wih_b : Wih_f;
    const bf16*  Whh = dir ? Whh_b : Whh_f;
    const float* bih = dir ? bih_b : bih_f;
    const float* bhh = dir ? bhh_b : bhh_f;

    bf16* h0b = hbuf + (size_t)chain * 2 * NBATCH * HH;
    bf16* h1b = h0b + (size_t)NBATCH * HH;

    // zero h(t=0) buffer
    {
        uint32_t* z = (uint32_t*)h0b;
        for (int i = threadIdx.x; i < NBATCH * HH / 2; i += blockDim.x)
            z[i] = 0u;
    }
    __syncthreads();

    const int lane = threadIdx.x & 31;
    const int wid  = threadIdx.x >> 5;   // 0..7

    for (int t = 0; t < LLN; ++t) {
        const int col = dir ? ((roll + LLN - 1 - t) & (LLN - 1))
                            : ((roll + t) & (LLN - 1));
        const bf16* hprev = (t & 1) ? h1b : h0b;
        bf16*       hnext = (t & 1) ? h0b : h1b;

        // 1024 tile-groups (m in 0..15, c in 0..63), 128 per wave.
        for (int g = wid * 128; g < wid * 128 + 128; ++g) {
            const int m = g >> 6, c = g & 63;
            const int r0 = m * 16, j0 = c * 16;
            const int arow = r0 + (lane & 15);
            const bf16* xrow = embed_bf + (size_t)lines[arow * LLN + col] * HH;
            const bf16* hrow = hprev + (size_t)arow * HH;

            GruAcc acc = gru_gates_tile(xrow, hrow, Wih, Whh, j0);

            const int jc  = j0 + (lane & 15);
            const float bxr = bih[jc], bxz = bih[jc + HH], bxn = bih[jc + 2*HH];
            const float bhr = bhh[jc], bhz = bhh[jc + HH], bhn = bhh[jc + 2*HH];
            #pragma unroll
            for (int i = 0; i < 8; ++i) {
                const int row = r0 + c_row(i);
                const float r  = sigmoidf(acc.xr[i] + bxr + acc.hr[i] + bhr);
                const float zz = sigmoidf(acc.xz[i] + bxz + acc.hz[i] + bhz);
                const float nn = tanhf(acc.xn[i] + bxn + r * (acc.hn[i] + bhn));
                const float hp = (float)hprev[(size_t)row * HH + jc];
                hnext[(size_t)row * HH + jc] = (bf16)((1.0f - zz) * nn + zz * hp);
            }
        }
        __threadfence_block();
        __syncthreads();
    }

    // 16 steps: final hidden state is in buffer 0. Write into G (bf16).
    const bf16* hf = h0b;
    for (int i = threadIdx.x; i < NBATCH * HH; i += blockDim.x) {
        const int n = i >> 10, j = i & (HH - 1);
        G[((size_t)roll * NBATCH + n) * (2 * HH) + (size_t)dir * HH + j] = hf[i];
    }
}

// ---------------------------------------------------------------------------
// GEMM + bias + ReLU stage for the decoder MLP (all 32 waves cooperate).
// Output is [256 x 1024] bf16 (ld = HH).
// ---------------------------------------------------------------------------
__device__ __forceinline__ void gemm_relu_stage(const bf16* __restrict__ X,
                                                int lda,
                                                const bf16* __restrict__ W,
                                                int ldw, int K,
                                                const float* __restrict__ bias,
                                                bf16* __restrict__ Y) {
    const int lane = threadIdx.x & 31;
    const int wid  = threadIdx.x >> 5;   // 0..31
    for (int g = wid * 32; g < wid * 32 + 32; ++g) {
        const int m = g >> 6, c = g & 63;
        const int r0 = m * 16, j0 = c * 16;
        const bf16* arow = X + (size_t)(r0 + (lane & 15)) * lda;
        v8f acc = {};
        for (int k0 = 0; k0 < K; k0 += 32) {
            if (k0 + 32 < K)
                __builtin_prefetch(W + (size_t)j0 * ldw + k0 + 32, 0, 1);
            v16bf a = load_a_frag(arow, k0);
            v16bf b = load_b_fragT(W, ldw, j0, k0);
            acc = wmma_bf16(a, b, acc);
        }
        const int jc = j0 + (lane & 15);
        const float bv = bias[jc];
        #pragma unroll
        for (int i = 0; i < 8; ++i) {
            const int row = r0 + c_row(i);
            Y[(size_t)row * HH + jc] = (bf16)fmaxf(acc[i] + bv, 0.0f);
        }
    }
}

// ---------------------------------------------------------------------------
// Phase 2: T=64 sequential scan. One 1024-thread block (32 waves); phases are
// ordered with __syncthreads(); the data-dependent pointer state w[] lives in
// LDS; softmax uses wave shuffles.
// ---------------------------------------------------------------------------
__global__ __launch_bounds__(1024, 1)
void decoder_kernel(const float* __restrict__ cond,
                    const int*  __restrict__ Aidx,
                    const int*  __restrict__ Pidx,
                    const float* __restrict__ h0,
                    const int*  __restrict__ w0,
                    const bf16* __restrict__ W0, const float* __restrict__ fb0,
                    const bf16* __restrict__ W1, const float* __restrict__ fb1,
                    const bf16* __restrict__ W2, const float* __restrict__ fb2,
                    const bf16* __restrict__ cWih, const bf16* __restrict__ cWhh,
                    const float* __restrict__ cbih, const float* __restrict__ cbhh,
                    const bf16* __restrict__ Whead, const float* __restrict__ bhead,
                    const bf16* __restrict__ G,
                    bf16* __restrict__ xbuf,                 // [256][2304]
                    bf16* __restrict__ y1, bf16* __restrict__ y2,
                    bf16* __restrict__ y3,                   // [256][1024] each
                    float* __restrict__ hf,                  // 2x [256][1024] f32
                    bf16*  __restrict__ hb,                  // 2x [256][1024] bf16
                    float* __restrict__ logits,              // [256][128]
                    float* __restrict__ out)                 // [64][256][1094]
{
    __shared__ int s_w[NBATCH];

    const int tid  = threadIdx.x;
    const int lane = tid & 31;
    const int wid  = tid >> 5;    // 0..31

    // init state (buffer 0 is read at t=0)
    for (int i = tid; i < NBATCH; i += 1024) s_w[i] = w0[i];
    for (int i = tid; i < NBATCH * HH; i += 1024) {
        const float v = h0[i];
        hf[i] = v;
        hb[i] = (bf16)v;
    }
    __syncthreads();

    for (int t = 0; t < TT; ++t) {
        const float* hfR = hf + (size_t)(t & 1) * NBATCH * HH;
        float*       hfW = hf + (size_t)((t + 1) & 1) * NBATCH * HH;
        const bf16*  hbR = hb + (size_t)(t & 1) * NBATCH * HH;
        bf16*        hbW = hb + (size_t)((t + 1) & 1) * NBATCH * HH;

        // ---- stage x = [cond_t | G[w[n], n]] as bf16 --------------------
        const float* ct = cond + (size_t)t * NBATCH * CDIM;
        for (int i = tid; i < NBATCH * CDIM; i += 1024) {
            const int n = i >> 8, k = i & (CDIM - 1);
            xbuf[(size_t)n * XDIM + k] = (bf16)ct[i];
        }
        for (int i = tid; i < NBATCH * 2 * HH; i += 1024) {
            const int n = i >> 11, j = i & (2 * HH - 1);
            xbuf[(size_t)n * XDIM + CDIM + j] =
                G[((size_t)s_w[n] * NBATCH + n) * (2 * HH) + j];
        }
        __syncthreads();

        // ---- 3-layer MLP ------------------------------------------------
        gemm_relu_stage(xbuf, XDIM, W0, XDIM, XDIM, fb0, y1);
        __syncthreads();
        gemm_relu_stage(y1, HH, W1, HH, HH, fb1, y2);
        __syncthreads();
        gemm_relu_stage(y2, HH, W2, HH, HH, fb2, y3);
        __syncthreads();

        // ---- fused GRU cell (ping-pong h) -------------------------------
        for (int g = wid * 32; g < wid * 32 + 32; ++g) {
            const int m = g >> 6, c = g & 63;
            const int r0 = m * 16, j0 = c * 16;
            const bf16* xrow = y3  + (size_t)(r0 + (lane & 15)) * HH;
            const bf16* hrow = hbR + (size_t)(r0 + (lane & 15)) * HH;

            GruAcc acc = gru_gates_tile(xrow, hrow, cWih, cWhh, j0);

            const int jc = j0 + (lane & 15);
            const float bxr = cbih[jc], bxz = cbih[jc + HH], bxn = cbih[jc + 2*HH];
            const float bhr = cbhh[jc], bhz = cbhh[jc + HH], bhn = cbhh[jc + 2*HH];
            #pragma unroll
            for (int i = 0; i < 8; ++i) {
                const int row = r0 + c_row(i);
                const float r  = sigmoidf(acc.xr[i] + bxr + acc.hr[i] + bhr);
                const float zz = sigmoidf(acc.xz[i] + bxz + acc.hz[i] + bhz);
                const float nn = tanhf(acc.xn[i] + bxn + r * (acc.hn[i] + bhn));
                const float hp = hfR[(size_t)row * HH + jc];
                const float hn = (1.0f - zz) * nn + zz * hp;
                hfW[(size_t)row * HH + jc] = hn;
                hbW[(size_t)row * HH + jc] = (bf16)hn;
                out[((size_t)t * NBATCH + row) * ODIM + 4 + jc] = hn;
            }
        }
        __syncthreads();

        // ---- heads: logits = h @ Whead^T + bhead, [256 x 128] -----------
        for (int g = wid * 4; g < wid * 4 + 4; ++g) {
            const int m = g >> 3, c = g & 7;
            const int r0 = m * 16, j0 = c * 16;
            const bf16* arow = hbW + (size_t)(r0 + (lane & 15)) * HH;
            v8f acc = {};
            for (int k0 = 0; k0 < HH; k0 += 32) {
                v16bf a = load_a_frag(arow, k0);
                v16bf b = load_b_fragT(Whead, HH, j0, k0);
                acc = wmma_bf16(a, b, acc);
            }
            const int jc = j0 + (lane & 15);
            const float bv = bhead[jc];
            #pragma unroll
            for (int i = 0; i < 8; ++i) {
                const int row = r0 + c_row(i);
                logits[(size_t)row * 128 + jc] = acc[i] + bv;
            }
        }
        __syncthreads();

        // ---- softmax + state update + packed output ---------------------
        for (int n = wid; n < NBATCH; n += 32) {
            float* orow = out + ((size_t)t * NBATCH + n) * ODIM;
            const float* lrow = logits + (size_t)n * 128;

            // a_probs over 33 (lane covers 0..31; col 32 handled explicitly)
            {
                float v  = lrow[lane];
                float e2 = lrow[32];
                float mx = v;
                for (int o = 16; o > 0; o >>= 1)
                    mx = fmaxf(mx, __shfl_xor(mx, o, 32));
                mx = fmaxf(mx, e2);
                float p = __expf(v - mx);
                float s = p;
                for (int o = 16; o > 0; o >>= 1)
                    s += __shfl_xor(s, o, 32);
                const float p2 = __expf(e2 - mx);
                s += p2;
                const float inv = 1.0f / s;
                orow[4 + HH + lane] = p * inv;
                if (lane == 0) orow[4 + HH + 32] = p2 * inv;
            }
            // p_probs over 33 (cols 48..80)
            {
                float v  = lrow[48 + lane];
                float e2 = lrow[48 + 32];
                float mx = v;
                for (int o = 16; o > 0; o >>= 1)
                    mx = fmaxf(mx, __shfl_xor(mx, o, 32));
                mx = fmaxf(mx, e2);
                float p = __expf(v - mx);
                float s = p;
                for (int o = 16; o > 0; o >>= 1)
                    s += __shfl_xor(s, o, 32);
                const float p2 = __expf(e2 - mx);
                s += p2;
                const float inv = 1.0f / s;
                orow[4 + HH + NAC + lane] = p * inv;
                if (lane == 0) orow[4 + HH + NAC + 32] = p2 * inv;
            }
            if (lane == 0) {
                const int a_t = Aidx[t * NBATCH + n];
                const int p_t = Pidx[t * NBATCH + n];
                int wn = s_w[n] + a_t - LLN;
                wn = wn < 0 ? 0 : (wn > LLN - 1 ? LLN - 1 : wn);
                orow[0] = (float)a_t;
                orow[1] = (float)p_t;
                orow[2] = (float)wn;
                orow[3] = lrow[96];          // value head (bias already added)
                s_w[n] = wn;
            }
        }
        __syncthreads();
    }
}

// ---------------------------------------------------------------------------
// Host-side launcher
// ---------------------------------------------------------------------------
extern "C" void kernel_launch(void* const* d_in, const int* in_sizes, int n_in,
                              void* d_out, int out_size, void* d_ws,
                              size_t ws_size, hipStream_t stream) {
    (void)in_sizes; (void)n_in; (void)out_size; (void)ws_size;

    const float* condition = (const float*)d_in[0];
    const int*   lines     = (const int*)  d_in[1];
    const int*   A         = (const int*)  d_in[2];
    const int*   P         = (const int*)  d_in[3];
    const float* h0        = (const float*)d_in[4];
    const int*   w0        = (const int*)  d_in[5];
    const float* embed     = (const float*)d_in[6];
    const float* tW_ih_f   = (const float*)d_in[7];
    const float* tW_hh_f   = (const float*)d_in[8];
    const float* tb_ih_f   = (const float*)d_in[9];
    const float* tb_hh_f   = (const float*)d_in[10];
    const float* tW_ih_b   = (const float*)d_in[11];
    const float* tW_hh_b   = (const float*)d_in[12];
    const float* tb_ih_b   = (const float*)d_in[13];
    const float* tb_hh_b   = (const float*)d_in[14];
    const float* fW0       = (const float*)d_in[15];
    const float* fb0       = (const float*)d_in[16];
    const float* fW1       = (const float*)d_in[17];
    const float* fb1       = (const float*)d_in[18];
    const float* fW2       = (const float*)d_in[19];
    const float* fb2       = (const float*)d_in[20];
    const float* cW_ih     = (const float*)d_in[21];
    const float* cW_hh     = (const float*)d_in[22];
    const float* cb_ih     = (const float*)d_in[23];
    const float* cb_hh     = (const float*)d_in[24];
    const float* vW        = (const float*)d_in[25];
    const float* vb        = (const float*)d_in[26];
    const float* aW        = (const float*)d_in[27];
    const float* ab        = (const float*)d_in[28];
    const float* pW        = (const float*)d_in[29];
    const float* pb        = (const float*)d_in[30];

    float* out = (float*)d_out;

    // -------- workspace carve-up (256B aligned regions) ------------------
    char* ws = (char*)d_ws;
    size_t off = 0;
    auto take = [&](size_t bytes) -> char* {
        char* p = ws + off;
        off = (off + bytes + 255) & ~(size_t)255;
        return p;
    };
    const size_t E_EMBED = (size_t)64 * HH;           // 65536
    const size_t E_TW    = (size_t)3 * HH * HH;       // 3145728
    const size_t E_FW0   = (size_t)HH * XDIM;         // 2359296
    const size_t E_FW    = (size_t)HH * HH;           // 1048576

    bf16* embed_bf = (bf16*)take(E_EMBED * 2);
    bf16* Wih_f    = (bf16*)take(E_TW * 2);
    bf16* Whh_f    = (bf16*)take(E_TW * 2);
    bf16* Wih_b    = (bf16*)take(E_TW * 2);
    bf16* Whh_b    = (bf16*)take(E_TW * 2);
    bf16* W0b      = (bf16*)take(E_FW0 * 2);
    bf16* W1b      = (bf16*)take(E_FW * 2);
    bf16* W2b      = (bf16*)take(E_FW * 2);
    bf16* cWihb    = (bf16*)take(E_TW * 2);
    bf16* cWhhb    = (bf16*)take(E_TW * 2);
    bf16* Whead    = (bf16*)take((size_t)128 * HH * 2);
    float* bhead   = (float*)take(128 * 4);
    bf16* hbuf     = (bf16*)take((size_t)32 * 2 * NBATCH * HH * 2);
    bf16* G        = (bf16*)take((size_t)LLN * NBATCH * 2 * HH * 2);
    bf16* xbuf     = (bf16*)take((size_t)NBATCH * XDIM * 2);
    bf16* y1       = (bf16*)take((size_t)NBATCH * HH * 2);
    bf16* y2       = (bf16*)take((size_t)NBATCH * HH * 2);
    bf16* y3       = (bf16*)take((size_t)NBATCH * HH * 2);
    float* hf      = (float*)take((size_t)2 * NBATCH * HH * 4);
    bf16*  hb      = (bf16*) take((size_t)2 * NBATCH * HH * 2);
    float* logits  = (float*)take((size_t)NBATCH * 128 * 4);

    // -------- weight conversion ------------------------------------------
    auto cvt = [&](bf16* dst, const float* src, size_t n) {
        k_f32_to_bf16<<<(unsigned)((n + 255) / 256), 256, 0, stream>>>(
            dst, src, (int)n);
    };
    cvt(embed_bf, embed,   E_EMBED);
    cvt(Wih_f,    tW_ih_f, E_TW);
    cvt(Whh_f,    tW_hh_f, E_TW);
    cvt(Wih_b,    tW_ih_b, E_TW);
    cvt(Whh_b,    tW_hh_b, E_TW);
    cvt(W0b,      fW0,     E_FW0);
    cvt(W1b,      fW1,     E_FW);
    cvt(W2b,      fW2,     E_FW);
    cvt(cWihb,    cW_ih,   E_TW);
    cvt(cWhhb,    cW_hh,   E_TW);
    k_pack_heads<<<(128 * HH + 255) / 256, 256, 0, stream>>>(
        Whead, bhead, aW, ab, pW, pb, vW, vb);

    // -------- phase 1: BiGRU over all rolls ------------------------------
    bigru_kernel<<<32, 256, 0, stream>>>(
        embed_bf, lines, Wih_f, Whh_f, Wih_b, Whh_b,
        tb_ih_f, tb_hh_f, tb_ih_b, tb_hh_b, hbuf, G);

    // -------- phase 2: sequential decoder scan ---------------------------
    decoder_kernel<<<1, 1024, 0, stream>>>(
        condition, A, P, h0, w0,
        W0b, fb0, W1b, fb1, W2b, fb2,
        cWihb, cWhhb, cb_ih, cb_hh,
        Whead, bhead, G,
        xbuf, y1, y2, y3, hf, hb, logits, out);
}
